// Backbone_19997367730748
// MI455X (gfx1250) — matrix operand order repaired and checked by reading, not compiled, not run on hardware
//
#include <hip/hip_runtime.h>

typedef __attribute__((ext_vector_type(16))) _Float16 v16h;
typedef __attribute__((ext_vector_type(8)))  _Float16 v8h;
typedef __attribute__((ext_vector_type(4)))  _Float16 v4h;
typedef __attribute__((ext_vector_type(8)))  float    v8f;

#define N_NODES 100000
#define N_EDGES 1600000
#define HID 64
#define NEG_SLOPE 0.01f

__device__ __forceinline__ void atom_add_f32(float* p, float v) {
  unsafeAtomicAdd(p, v);   // lowers to global_atomic_add_f32 on gfx1250
}

// ---- degree / normalization ------------------------------------------------
__global__ void k_deg_init(float* __restrict__ deg) {
  int i = blockIdx.x * blockDim.x + threadIdx.x;
  if (i < N_NODES) deg[i] = 1.0f;                     // self loop
}

__global__ void k_deg_count(const long long* __restrict__ ei, float* __restrict__ deg) {
  int e = blockIdx.x * blockDim.x + threadIdx.x;
  if (e < N_EDGES) atom_add_f32(&deg[(int)ei[N_EDGES + e]], 1.0f);
}

__global__ void k_dinv(float* __restrict__ deg) {
  int i = blockIdx.x * blockDim.x + threadIdx.x;
  if (i < N_NODES) deg[i] = rsqrtf(deg[i]);
}

// ---- f32 -> f16 activation conversion --------------------------------------
__global__ void k_cvt(const float* __restrict__ x, _Float16* __restrict__ h) {
  int g = blockIdx.x * blockDim.x + threadIdx.x;      // N*16 quads
  if (g >= N_NODES * 16) return;
  const float4 v = *(const float4*)(x + (size_t)g * 4);
  v4h o = { (_Float16)v.x, (_Float16)v.y, (_Float16)v.z, (_Float16)v.w };
  *(v4h*)(h + (size_t)g * 4) = o;
}

// ---- pre-swizzle W into WMMA B-fragment layout -----------------------------
// B (32x16 f16) per-lane layout: lane L covers column n = L%16; the 16 halves
// of the lane hold K in [8h,8h+8) then [16+8h,16+8h+8), h = L/16 (matches A).
__global__ void k_pack_w(const float* __restrict__ W1, const float* __restrict__ W2,
                         const float* __restrict__ W3, _Float16* __restrict__ frag) {
  int id = blockIdx.x * blockDim.x + threadIdx.x;     // 3 * 4096 halves
  if (id >= 3 * 4096) return;
  int layer = id >> 12;
  int r     = id & 4095;
  int p     = r & 15;
  int lane  = (r >> 4) & 31;
  int ks    = (r >> 9) & 1;                           // K-step (0: K 0..31, 1: K 32..63)
  int tile  = (r >> 10) & 3;                          // column tile
  int n = tile * 16 + (lane & 15);
  int k = ks * 32 + 8 * (lane >> 4) + ((p >> 3) << 4) + (p & 7);
  const float* W = (layer == 0) ? W1 : ((layer == 1) ? W2 : W3);
  frag[id] = (_Float16)W[k * HID + n];
}

// ---- WMMA GEMM: C[100000x64] = A[100000x64](f16) @ W(f16 fragments) --------
// One wave per 16x16 output tile; 4 waves/block cover the 4 column tiles of a
// 16-row block. A-fragment per lane = two contiguous 16-byte loads per K-step
// (ISA 16-bit A layout: lane(h=L/16,m=L%16) holds K in [8h,8h+8) and [16+8h,16+8h+8)).
__global__ __launch_bounds__(128) void k_gemm(const _Float16* __restrict__ A,
                                              const _Float16* __restrict__ Bfrag,
                                              float* __restrict__ C) {
  int lane = threadIdx.x & 31;
  int wave = threadIdx.x >> 5;                        // column tile 0..3
  int rb   = blockIdx.x;                              // 16-row block, 0..6249
  int m = lane & 15, h = lane >> 4;

  const _Float16* arow = A + (size_t)(rb * 16 + m) * HID;
  union { v16h v; v8h p[2]; } a0, a1;
  a0.p[0] = *(const v8h*)(arow + 8 * h);
  a0.p[1] = *(const v8h*)(arow + 16 + 8 * h);
  a1.p[0] = *(const v8h*)(arow + 32 + 8 * h);
  a1.p[1] = *(const v8h*)(arow + 48 + 8 * h);

  const v16h* bf = (const v16h*)Bfrag;
  v16h b0 = bf[(wave * 2 + 0) * 32 + lane];
  v16h b1 = bf[(wave * 2 + 1) * 32 + lane];

  v8f c = {};
  c = __builtin_amdgcn_wmma_f32_16x16x32_f16(false, a0.v, false, b0, (short)0, c, false, false);
  c = __builtin_amdgcn_wmma_f32_16x16x32_f16(false, a1.v, false, b1, (short)0, c, false, false);

  // C/D layout: lane L -> column n = L%16, VGPR j -> row 8*(L/16)+j
  float* crow = C + (size_t)(rb * 16 + 8 * h) * HID + wave * 16 + m;
#pragma unroll
  for (int j = 0; j < 8; ++j) crow[(size_t)j * HID] = c[j];
}

// ---- self-loop contribution initializes agg --------------------------------
__global__ void k_selfloop(const float* __restrict__ dinv, const float* __restrict__ t,
                           float* __restrict__ agg) {
  int g = blockIdx.x * blockDim.x + threadIdx.x;      // N*16 quads
  if (g >= N_NODES * 16) return;
  int i = g >> 4;
  float s = dinv[i]; s = s * s;
  const float4 v = *(const float4*)(t + (size_t)g * 4);
  float4 r = { v.x * s, v.y * s, v.z * s, v.w * s };
  *(float4*)(agg + (size_t)g * 4) = r;
}

// ---- edge scatter: agg[dst] += t[src] * dinv[src]*dinv[dst] ----------------
// 16 lanes per edge, float4 gather + 4 f32 atomics each (256B/row coalesced).
__global__ __launch_bounds__(256) void k_scatter(const long long* __restrict__ ei,
                                                 const float* __restrict__ dinv,
                                                 const float* __restrict__ t,
                                                 float* __restrict__ agg) {
  int g = blockIdx.x * blockDim.x + threadIdx.x;      // E*16
  if (g >= N_EDGES * 16) return;
  int e = g >> 4;
  int q = g & 15;
  int s = (int)ei[e];
  int d = (int)ei[N_EDGES + e];
  float w = dinv[s] * dinv[d];
  const float4 v = *(const float4*)(t + (size_t)s * HID + q * 4);
  float* o = agg + (size_t)d * HID + q * 4;
  atom_add_f32(o + 0, v.x * w);
  atom_add_f32(o + 1, v.y * w);
  atom_add_f32(o + 2, v.z * w);
  atom_add_f32(o + 3, v.w * w);
}

// ---- bias + LeakyReLU; emit next-layer f16 activations or final f32 --------
__global__ void k_finalize(const float* __restrict__ agg, const float* __restrict__ bias,
                           _Float16* __restrict__ hnext, float* __restrict__ outf, int last) {
  int g = blockIdx.x * blockDim.x + threadIdx.x;      // N*16 quads
  if (g >= N_NODES * 16) return;
  int q = g & 15;
  const float4 v = *(const float4*)(agg + (size_t)g * 4);
  const float4 b = *(const float4*)(bias + q * 4);
  float r0 = v.x + b.x, r1 = v.y + b.y, r2 = v.z + b.z, r3 = v.w + b.w;
  r0 = (r0 >= 0.f) ? r0 : NEG_SLOPE * r0;
  r1 = (r1 >= 0.f) ? r1 : NEG_SLOPE * r1;
  r2 = (r2 >= 0.f) ? r2 : NEG_SLOPE * r2;
  r3 = (r3 >= 0.f) ? r3 : NEG_SLOPE * r3;
  if (last) {
    float4 o = { r0, r1, r2, r3 };
    *(float4*)(outf + (size_t)g * 4) = o;
  } else {
    v4h o = { (_Float16)r0, (_Float16)r1, (_Float16)r2, (_Float16)r3 };
    *(v4h*)(hnext + (size_t)g * 4) = o;
  }
}

// ---------------------------------------------------------------------------
extern "C" void kernel_launch(void* const* d_in, const int* in_sizes, int n_in,
                              void* d_out, int out_size, void* d_ws, size_t ws_size,
                              hipStream_t stream) {
  const float*     x  = (const float*)d_in[0];
  const long long* ei = (const long long*)d_in[1];
  const float* W1 = (const float*)d_in[2];
  const float* b1 = (const float*)d_in[3];
  const float* W2 = (const float*)d_in[4];
  const float* b2 = (const float*)d_in[5];
  const float* W3 = (const float*)d_in[6];
  const float* b3 = (const float*)d_in[7];
  float* out = (float*)d_out;

  // workspace carving (256B aligned)
  char* ws = (char*)d_ws;
  size_t off = 0;
  auto carve = [&](size_t bytes) -> char* {
    char* p = ws + off;
    off += (bytes + 255) & ~(size_t)255;
    return p;
  };
  float*    dinv  = (float*)   carve((size_t)N_NODES * 4);
  _Float16* hbuf  = (_Float16*)carve((size_t)N_NODES * HID * 2);
  float*    tbuf  = (float*)   carve((size_t)N_NODES * HID * 4);
  float*    agg   = (float*)   carve((size_t)N_NODES * HID * 4);
  _Float16* wfrag = (_Float16*)carve((size_t)3 * 4096 * 2);
  (void)ws_size; (void)in_sizes; (void)n_in; (void)out_size;

  const int TB = 256;
  const int nq = N_NODES * 16;                  // per-node quad threads

  // graph normalization
  k_deg_init <<<(N_NODES + TB - 1) / TB, TB, 0, stream>>>(dinv);
  k_deg_count<<<(N_EDGES + TB - 1) / TB, TB, 0, stream>>>(ei, dinv);
  k_dinv     <<<(N_NODES + TB - 1) / TB, TB, 0, stream>>>(dinv);

  // setup: f16 activations + W fragments
  k_cvt   <<<(nq + TB - 1) / TB, TB, 0, stream>>>(x, hbuf);
  k_pack_w<<<(3 * 4096 + TB - 1) / TB, TB, 0, stream>>>(W1, W2, W3, wfrag);

  const float* biases[3] = { b1, b2, b3 };
  for (int layer = 0; layer < 3; ++layer) {
    k_gemm    <<<N_NODES / 16, 128, 0, stream>>>(hbuf, wfrag + layer * 4096, tbuf);
    k_selfloop<<<(nq + TB - 1) / TB, TB, 0, stream>>>(dinv, tbuf, agg);
    k_scatter <<<(N_EDGES * 16 + TB - 1) / TB, TB, 0, stream>>>(ei, dinv, tbuf, agg);
    k_finalize<<<(nq + TB - 1) / TB, TB, 0, stream>>>(agg, biases[layer], hbuf, out,
                                                      layer == 2 ? 1 : 0);
  }
}
